// Seq2SeqGru_52518860095510
// MI455X (gfx1250) — matrix operand order repaired and checked by reading, not compiled, not run on hardware
//
#include <hip/hip_runtime.h>
#include <hip/hip_bf16.h>

// ---------------------------------------------------------------------------
// Types for CDNA5 WMMA
// ---------------------------------------------------------------------------
typedef __bf16 bf16;
typedef bf16  v16bf __attribute__((ext_vector_type(16)));
typedef bf16  v8bf  __attribute__((ext_vector_type(8)));
typedef float v8f   __attribute__((ext_vector_type(8)));

#define HID   512
#define BATCH 32
#define TENC  256
#define TDEC  32
#define VOC   50000
#define G3    1536   // 3*HID
#define VTILES (VOC / 16)   // 3125

__device__ __forceinline__ bf16 f2bf(float f) {
  union { float f; unsigned u; } x; x.f = f;
  unsigned r = x.u + 0x7FFFu + ((x.u >> 16) & 1u);   // round-to-nearest-even
  unsigned short h = (unsigned short)(r >> 16);
  return __builtin_bit_cast(bf16, h);
}

__device__ __forceinline__ float sigmoidf_(float x) {
  return 1.0f / (1.0f + __expf(-x));
}

// A-fragment (16x32 bf16, M x K): lane l -> row l&15,
// halves at kb + (l>=16 ? 8:0) and kb+16 + (l>=16 ? 8:0)
__device__ __forceinline__ v16bf load_a_frag(const bf16* __restrict__ base,
                                             int ld, int row, int kb, int lane) {
  const bf16* p = base + (size_t)row * ld + kb + ((lane >> 4) << 3);
  v8bf lo = *(const v8bf*)(p);
  v8bf hi = *(const v8bf*)(p + 16);
  v16bf r;
#pragma unroll
  for (int i = 0; i < 8; ++i) { r[i] = lo[i]; r[i + 8] = hi[i]; }
  return r;
}

// B-fragment (32x16 bf16, K x N) with B supplied as rows-per-output-column
// (row j of `base` = column j of B): lane n -> memory row n&15,
// halves at kb + (n>=16 ? 8:0) and kb+16 + (n>=16 ? 8:0)
__device__ __forceinline__ v16bf load_b_frag(const bf16* __restrict__ base,
                                             int ld, int row, int kb, int lane) {
  const bf16* p = base + (size_t)row * ld + kb + ((lane >> 4) << 3);
  v8bf lo = *(const v8bf*)(p);
  v8bf hi = *(const v8bf*)(p + 16);
  v16bf r;
#pragma unroll
  for (int i = 0; i < 8; ++i) { r[i] = lo[i]; r[i + 8] = hi[i]; }
  return r;
}

// ---------------------------------------------------------------------------
// GRU step pieces (shared by persistent encoder and per-step decoder kernels)
// Block = 384 threads = 12 waves; each wave computes one 16x16 tile of
// gh = h @ Whh^T for (gate g, batch-half, 16 hidden cols). Each WG owns 32
// hidden units (blockIdx.x*32 .. +31) across all three gates.
// ---------------------------------------------------------------------------
__device__ __forceinline__ void gru_tiles(const bf16* __restrict__ hB,
                                          const bf16* __restrict__ Whh,
                                          float* __restrict__ gh_s /*3*32*32*/) {
  const int lane = threadIdx.x & 31;
  const int w    = threadIdx.x >> 5;     // 0..11
  const int g    = w >> 2;               // gate 0..2
  const int sub  = w & 3;
  const int b0   = (sub & 1) << 4;       // 0 or 16
  const int jt   = sub >> 1;             // 0..1
  const int j0   = blockIdx.x * 32 + jt * 16;

  v8f acc = {};
  const int arow = b0 + (lane & 15);
  const int brow = g * HID + j0 + (lane & 15);
#pragma unroll 4
  for (int kb = 0; kb < HID; kb += 32) {
    v16bf a = load_a_frag(hB,  HID, arow, kb, lane);
    v16bf b = load_b_frag(Whh, HID, brow, kb, lane);
    acc = __builtin_amdgcn_wmma_f32_16x16x32_bf16(false, a, false, b,
                                                  (short)0, acc, false, false);
  }
  const int nn = lane & 15;
  const int mb = (lane >> 4) << 3;
#pragma unroll
  for (int v = 0; v < 8; ++v) {
    int m  = mb + v;                     // 0..15
    int bb = b0 + m;                     // batch row
    int jj = jt * 16 + nn;               // 0..31 within WG slice
    gh_s[g * (BATCH * 32) + bb * 32 + jj] = acc[v];
  }
}

__device__ __forceinline__ void gru_update(const float* __restrict__ gh_s,
                                           const float* __restrict__ GX, int rowbase,
                                           const float* __restrict__ bhh,
                                           const float* __restrict__ hF_rd,
                                           float* __restrict__ hF_wr,
                                           bf16* __restrict__ hB_wr) {
  const int j0 = blockIdx.x * 32;
  for (int e = threadIdx.x; e < BATCH * 32; e += blockDim.x) {
    int b  = e >> 5;
    int jj = e & 31;
    int j  = j0 + jj;
    float ghr = gh_s[0 * (BATCH * 32) + b * 32 + jj] + bhh[j];
    float ghz = gh_s[1 * (BATCH * 32) + b * 32 + jj] + bhh[HID + j];
    float ghn = gh_s[2 * (BATCH * 32) + b * 32 + jj] + bhh[2 * HID + j];
    const float* gx = GX + (size_t)(rowbase + b) * G3;  // bih already folded in
    float r = sigmoidf_(gx[j] + ghr);
    float z = sigmoidf_(gx[HID + j] + ghz);
    float n = tanhf(gx[2 * HID + j] + r * ghn);
    float ho = hF_rd[b * HID + j];
    float hn = (1.0f - z) * n + z * ho;
    hF_wr[b * HID + j] = hn;
    hB_wr[b * HID + j] = f2bf(hn);
  }
}

// ---------------------------------------------------------------------------
// Persistent encoder: 16 WGs, 256 sequential GRU steps, device-scope barrier.
// h double-buffered (ping-pong on t parity); final h lands in buffer 0.
// ---------------------------------------------------------------------------
__global__ __launch_bounds__(384)
void encoder_scan_kernel(const float* __restrict__ GXe,
                         const bf16*  __restrict__ WhhE,
                         const float* __restrict__ bhhE,
                         float* __restrict__ hF, bf16* __restrict__ hB,
                         int* __restrict__ bar) {
  __shared__ float gh_s[3 * BATCH * 32];
  const int HSZ = BATCH * HID;
  for (int t = 0; t < TENC; ++t) {
    const int rd = t & 1, wr = (t + 1) & 1;
    // prefetch next step's gx slab into cache while we compute
    if (t + 1 < TENC) {
      const float* nxt = GXe + (size_t)((t + 1) * BATCH) * G3
                       + blockIdx.x * 32 + (threadIdx.x & 31);
      __builtin_prefetch(nxt, 0, 1);
    }
    gru_tiles(hB + (size_t)rd * HSZ, WhhE, gh_s);
    __syncthreads();
    gru_update(gh_s, GXe, t * BATCH, bhhE,
               hF + (size_t)rd * HSZ, hF + (size_t)wr * HSZ, hB + (size_t)wr * HSZ);
    __threadfence();
    __syncthreads();
    if (threadIdx.x == 0) {
      __hip_atomic_fetch_add(bar, 1, __ATOMIC_RELEASE, __HIP_MEMORY_SCOPE_AGENT);
      const int target = gridDim.x * (t + 1);
      while (__hip_atomic_load(bar, __ATOMIC_ACQUIRE, __HIP_MEMORY_SCOPE_AGENT) < target)
        __builtin_amdgcn_s_sleep(1);
    }
    __syncthreads();
  }
}

// One decoder GRU step (kernel boundary is the global sync): 16 WGs x 384
__global__ __launch_bounds__(384)
void dec_gru_kernel(const float* __restrict__ GXd,
                    const bf16*  __restrict__ WhhD,
                    const float* __restrict__ bhhD,
                    const float* __restrict__ hF_rd, float* __restrict__ hF_wr,
                    const bf16*  __restrict__ hB_rd, bf16* __restrict__ hB_wr,
                    int t) {
  __shared__ float gh_s[3 * BATCH * 32];
  gru_tiles(hB_rd, WhhD, gh_s);
  __syncthreads();
  gru_update(gh_s, GXd, t * BATCH, bhhD, hF_rd, hF_wr, hB_wr);
}

// ---------------------------------------------------------------------------
// Parallel GEMM: GX[R,1536] = X[R,512](bf16) @ W[1536,512](bf16, row=out col)^T + bih
// Block 256 thr = 8 waves; wave = one 16x16 tile; WG tile = 32 rows x 64 cols.
// ---------------------------------------------------------------------------
__global__ __launch_bounds__(256)
void gemm_gx_kernel(const bf16* __restrict__ X, const bf16* __restrict__ W,
                    const float* __restrict__ bih, float* __restrict__ GX) {
  const int lane = threadIdx.x & 31;
  const int w    = threadIdx.x >> 5;
  const int b0   = blockIdx.x * 32 + ((w & 1) << 4);
  const int j0   = blockIdx.y * 64 + ((w >> 1) << 4);
  v8f acc = {};
  const int arow = b0 + (lane & 15);
  const int brow = j0 + (lane & 15);
#pragma unroll 4
  for (int kb = 0; kb < HID; kb += 32) {
    v16bf a = load_a_frag(X, HID, arow, kb, lane);
    v16bf b = load_b_frag(W, HID, brow, kb, lane);
    acc = __builtin_amdgcn_wmma_f32_16x16x32_bf16(false, a, false, b,
                                                  (short)0, acc, false, false);
  }
  const int nn = lane & 15;
  const int mb = (lane >> 4) << 3;
#pragma unroll
  for (int v = 0; v < 8; ++v) {
    int r = b0 + mb + v;
    int j = j0 + nn;
    GX[(size_t)r * G3 + j] = acc[v] + bih[j];
  }
}

// ---------------------------------------------------------------------------
// Logits: [32,50000] = hB[32,512] @ WgT[50000,512]^T + bg (WgT row v = Wg col v)
// Each wave owns TWO adjacent 16-col vocab tiles -> two independent WMMA
// accumulator chains (2x ILP) and shared A-fragment loads.
// WG = 8 waves = 2 batch-halves x 4 tile-pairs = 32 rows x 128 cols.
// ---------------------------------------------------------------------------
__global__ __launch_bounds__(256)
void logits_kernel(const bf16* __restrict__ hB, const bf16* __restrict__ WgT,
                   const float* __restrict__ bg, float* __restrict__ logits) {
  const int lane = threadIdx.x & 31;
  const int w    = threadIdx.x >> 5;
  const int b0   = (w & 1) << 4;
  const int vt0  = blockIdx.x * 8 + ((w >> 1) << 1);  // wave-uniform
  if (vt0 >= VTILES) return;                          // whole-wave exit
  const int  v0  = vt0 * 16;
  const bool two = (vt0 + 1) < VTILES;                // wave-uniform

  v8f acc0 = {}, acc1 = {};
  const int arow  = b0 + (lane & 15);
  const int brow0 = v0 + (lane & 15);

  if (two) {
#pragma unroll 4
    for (int kb = 0; kb < HID; kb += 32) {
      v16bf a   = load_a_frag(hB,  HID, arow,       kb, lane);
      v16bf bf0 = load_b_frag(WgT, HID, brow0,      kb, lane);
      v16bf bf1 = load_b_frag(WgT, HID, brow0 + 16, kb, lane);
      acc0 = __builtin_amdgcn_wmma_f32_16x16x32_bf16(false, a, false, bf0,
                                                     (short)0, acc0, false, false);
      acc1 = __builtin_amdgcn_wmma_f32_16x16x32_bf16(false, a, false, bf1,
                                                     (short)0, acc1, false, false);
    }
  } else {
#pragma unroll 4
    for (int kb = 0; kb < HID; kb += 32) {
      v16bf a   = load_a_frag(hB,  HID, arow,  kb, lane);
      v16bf bf0 = load_b_frag(WgT, HID, brow0, kb, lane);
      acc0 = __builtin_amdgcn_wmma_f32_16x16x32_bf16(false, a, false, bf0,
                                                     (short)0, acc0, false, false);
    }
  }

  const int nn = lane & 15;
  const int mb = (lane >> 4) << 3;
#pragma unroll
  for (int v = 0; v < 8; ++v) {
    int bb = b0 + mb + v;
    int vv = v0 + nn;
    logits[(size_t)bb * VOC + vv] = acc0[v] + bg[vv];
  }
  if (two) {
#pragma unroll
    for (int v = 0; v < 8; ++v) {
      int bb = b0 + mb + v;
      int vv = v0 + 16 + nn;
      logits[(size_t)bb * VOC + vv] = acc1[v] + bg[vv];
    }
  }
}

// Per-row log-softmax + NLL accumulate: one WG per batch row.
__global__ __launch_bounds__(256)
void loss_kernel(const float* __restrict__ logits, const int* __restrict__ qw,
                 int t, float* __restrict__ loss) {
  const int b = blockIdx.x;
  const float* row = logits + (size_t)b * VOC;
  __shared__ float red[256];
  float m = -1e30f;
  for (int v = threadIdx.x; v < VOC; v += 256) m = fmaxf(m, row[v]);
  red[threadIdx.x] = m; __syncthreads();
  for (int s = 128; s > 0; s >>= 1) {
    if (threadIdx.x < s) red[threadIdx.x] = fmaxf(red[threadIdx.x], red[threadIdx.x + s]);
    __syncthreads();
  }
  const float rmax = red[0]; __syncthreads();
  float sum = 0.0f;
  for (int v = threadIdx.x; v < VOC; v += 256) sum += __expf(row[v] - rmax);
  red[threadIdx.x] = sum; __syncthreads();
  for (int s = 128; s > 0; s >>= 1) {
    if (threadIdx.x < s) red[threadIdx.x] += red[threadIdx.x + s];
    __syncthreads();
  }
  if (threadIdx.x == 0) {
    int tgt = qw[b * TDEC + t];
    float lp = row[tgt] - rmax - __logf(red[0]);
    atomicAdd(loss, -lp);
  }
}

// ---------------------------------------------------------------------------
// Prep kernels
// ---------------------------------------------------------------------------
__global__ void init_kernel(float* hF, bf16* hB, int* bar, float* loss) {
  int i = blockIdx.x * blockDim.x + threadIdx.x;
  if (i == 0) { *bar = 0; *loss = 0.0f; }
  if (i < 2 * BATCH * HID) { hF[i] = 0.0f; hB[i] = f2bf(0.0f); }
}

__global__ void f2bf_kernel(const float* __restrict__ src, bf16* __restrict__ dst, int n) {
  int i = blockIdx.x * blockDim.x + threadIdx.x;
  if (i < n) dst[i] = f2bf(src[i]);
}

// LDS-tiled transpose: WgT[v,k] = Wg[k,v]  (Wg: [512,50000], WgT: [50000,512] bf16)
__global__ __launch_bounds__(256)
void transpose_wg_kernel(const float* __restrict__ Wg, bf16* __restrict__ WgT) {
  __shared__ float tile[32][33];
  const int vb = blockIdx.x * 32;
  const int kb = blockIdx.y * 32;
  const int tx = threadIdx.x & 31;
  const int ty = threadIdx.x >> 5;   // 0..7
  for (int r = ty; r < 32; r += 8) {
    int k = kb + r, v = vb + tx;
    tile[r][tx] = (v < VOC) ? Wg[(size_t)k * VOC + v] : 0.0f;
  }
  __syncthreads();
  for (int r = ty; r < 32; r += 8) {
    int v = vb + r, k = kb + tx;
    if (v < VOC) WgT[(size_t)v * HID + k] = f2bf(tile[tx][r]);
  }
}

__global__ void embed_enc_kernel(const int* __restrict__ cw,
                                 const float* __restrict__ emb, bf16* __restrict__ X) {
  const int r = blockIdx.x;              // r = t*32 + b
  const int t = r >> 5, b = r & 31;
  const int idx = cw[b * TENC + t];
  const float* src = emb + (size_t)idx * HID;
  bf16* dst = X + (size_t)r * HID;
  for (int k = threadIdx.x; k < HID; k += blockDim.x) dst[k] = f2bf(src[k]);
}

__global__ void embed_dec_kernel(const int* __restrict__ qw,
                                 const float* __restrict__ emb, bf16* __restrict__ X) {
  const int r = blockIdx.x;              // r = t*32 + b
  const int t = r >> 5, b = r & 31;
  const int idx = (t == 0) ? 2 : qw[b * TDEC + (t - 1)];  // SOS then shifted tgt
  const float* src = emb + (size_t)idx * HID;
  bf16* dst = X + (size_t)r * HID;
  for (int k = threadIdx.x; k < HID; k += blockDim.x) {
    float f = src[k];
    dst[k] = f2bf(f > 0.0f ? f : 0.0f);  // relu
  }
}

__global__ void finish_kernel(const float* __restrict__ loss, float* __restrict__ out) {
  if (threadIdx.x == 0 && blockIdx.x == 0) out[0] = loss[0] / (float)TDEC;
}

// ---------------------------------------------------------------------------
// Host launcher
// ---------------------------------------------------------------------------
extern "C" void kernel_launch(void* const* d_in, const int* in_sizes, int n_in,
                              void* d_out, int out_size, void* d_ws, size_t ws_size,
                              hipStream_t stream) {
  (void)in_sizes; (void)n_in; (void)out_size; (void)ws_size;
  const int*   cw   = (const int*)  d_in[0];
  const int*   qw   = (const int*)  d_in[1];
  const float* embE = (const float*)d_in[2];
  const float* embD = (const float*)d_in[3];
  const float* WihE = (const float*)d_in[4];
  const float* WhhE = (const float*)d_in[5];
  const float* bihE = (const float*)d_in[6];
  const float* bhhE = (const float*)d_in[7];
  const float* WihD = (const float*)d_in[8];
  const float* WhhD = (const float*)d_in[9];
  const float* bihD = (const float*)d_in[10];
  const float* bhhD = (const float*)d_in[11];
  const float* Wg   = (const float*)d_in[12];
  const float* bg   = (const float*)d_in[13];
  float* out = (float*)d_out;

  char* ws = (char*)d_ws;
  size_t off = 0;
  auto alloc = [&](size_t bytes) {
    void* p = ws + off;
    off = (off + bytes + 255) & ~(size_t)255;
    return p;
  };
  const size_t WMAT = (size_t)G3 * HID;            // 786432
  bf16*  WihE_b = (bf16*) alloc(WMAT * 2);
  bf16*  WhhE_b = (bf16*) alloc(WMAT * 2);
  bf16*  WihD_b = (bf16*) alloc(WMAT * 2);
  bf16*  WhhD_b = (bf16*) alloc(WMAT * 2);
  bf16*  WgT_b  = (bf16*) alloc((size_t)VOC * HID * 2);
  bf16*  Xe_b   = (bf16*) alloc((size_t)TENC * BATCH * HID * 2);
  bf16*  Xd_b   = (bf16*) alloc((size_t)TDEC * BATCH * HID * 2);
  float* GXe    = (float*)alloc((size_t)TENC * BATCH * G3 * 4);
  float* GXd    = (float*)alloc((size_t)TDEC * BATCH * G3 * 4);
  float* logits = (float*)alloc((size_t)BATCH * VOC * 4);
  float* hF     = (float*)alloc((size_t)2 * BATCH * HID * 4);
  bf16*  hB     = (bf16*) alloc((size_t)2 * BATCH * HID * 2);
  float* loss   = (float*)alloc(256);
  int*   bar    = (int*)  alloc(256);
  const size_t HSZ = (size_t)BATCH * HID;

  // ---- prep ----
  init_kernel<<<dim3((2 * BATCH * HID + 255) / 256), dim3(256), 0, stream>>>(hF, hB, bar, loss);
  {
    dim3 g((unsigned)((WMAT + 255) / 256)), b(256);
    f2bf_kernel<<<g, b, 0, stream>>>(WihE, WihE_b, (int)WMAT);
    f2bf_kernel<<<g, b, 0, stream>>>(WhhE, WhhE_b, (int)WMAT);
    f2bf_kernel<<<g, b, 0, stream>>>(WihD, WihD_b, (int)WMAT);
    f2bf_kernel<<<g, b, 0, stream>>>(WhhD, WhhD_b, (int)WMAT);
  }
  transpose_wg_kernel<<<dim3((VOC + 31) / 32, HID / 32), dim3(256), 0, stream>>>(Wg, WgT_b);
  embed_enc_kernel<<<dim3(TENC * BATCH), dim3(128), 0, stream>>>(cw, embE, Xe_b);
  embed_dec_kernel<<<dim3(TDEC * BATCH), dim3(128), 0, stream>>>(qw, embD, Xd_b);

  // ---- parallel gx GEMMs (bih folded in) ----
  gemm_gx_kernel<<<dim3(TENC * BATCH / 32, G3 / 64), dim3(256), 0, stream>>>(Xe_b, WihE_b, bihE, GXe);
  gemm_gx_kernel<<<dim3(TDEC * BATCH / 32, G3 / 64), dim3(256), 0, stream>>>(Xd_b, WihD_b, bihD, GXd);

  // ---- persistent encoder scan (final h -> buffer 0) ----
  encoder_scan_kernel<<<dim3(HID / 32), dim3(384), 0, stream>>>(GXe, WhhE_b, bhhE, hF, hB, bar);

  // ---- decoder: GRU step, logits, loss per timestep ----
  for (int t = 0; t < TDEC; ++t) {
    const int rd = t & 1, wr = (t + 1) & 1;
    dec_gru_kernel<<<dim3(HID / 32), dim3(384), 0, stream>>>(
        GXd, WhhD_b, bhhD,
        hF + (size_t)rd * HSZ, hF + (size_t)wr * HSZ,
        hB + (size_t)rd * HSZ, hB + (size_t)wr * HSZ, t);
    logits_kernel<<<dim3((VTILES + 7) / 8), dim3(256), 0, stream>>>(
        hB + (size_t)wr * HSZ, WgT_b, bg, logits);
    loss_kernel<<<dim3(BATCH), dim3(256), 0, stream>>>(logits, qw, t, loss);
  }

  finish_kernel<<<dim3(1), dim3(1), 0, stream>>>(loss, out);
}
